// TrainedLiftGCNN_87557203296302
// MI455X (gfx1250) — matrix operand order repaired and checked by reading, not compiled, not run on hardware
//
#include <hip/hip_runtime.h>
#include <hip/hip_bf16.h>

typedef __attribute__((ext_vector_type(16))) __bf16 v16bf;
typedef __attribute__((ext_vector_type(8)))  float  v8f;
typedef unsigned short u16;
typedef unsigned int   u32;

#define PI_F 3.14159265358979f
#define COUT 128   // 16 channels * 8 orientations

__device__ __forceinline__ u16 f2bf(float f) {
  union { float f; u32 u; } v; v.f = f;
  u32 r = v.u + 0x7FFFu + ((v.u >> 16) & 1u);   // round-to-nearest-even
  return (u16)(r >> 16);
}

// ---------------- reflect pad (+optional BN+ReLU), f32 -> bf16 ----------------
__global__ void pad_bn_relu_bf16(const float* __restrict__ in, u16* __restrict__ outp,
                                 int C, int H, int W, int p,
                                 const float* __restrict__ st2,   // [co]{mean, invstd}
                                 const float* __restrict__ gamma,
                                 const float* __restrict__ beta,
                                 int doBN, int total) {
  int Hp = H + 2 * p, Wp = W + 2 * p;
  for (int idx = blockIdx.x * blockDim.x + threadIdx.x; idx < total;
       idx += gridDim.x * blockDim.x) {
    int x = idx % Wp;
    int t = idx / Wp;
    int y = t % Hp; t /= Hp;
    int c = t % C;  int b = t / C;
    int sy = y - p; if (sy < 0) sy = -sy; if (sy >= H) sy = 2 * H - 2 - sy;
    int sx = x - p; if (sx < 0) sx = -sx; if (sx >= W) sx = 2 * W - 2 - sx;
    float v = in[(((size_t)b * C + c) * H + sy) * W + sx];
    if (doBN) {
      int co = c >> 3;                       // channel layout c = co*8 + m
      v = gamma[co] * (v - st2[2 * co]) * st2[2 * co + 1] + beta[co];
      v = fmaxf(v, 0.f);
    }
    outp[idx] = f2bf(v);
  }
}

// ---------------- bilinear kernel rotation (matches _rotate_kernel) ----------------
__device__ __forceinline__ float rotate_sample(const float* __restrict__ src, int k,
                                               int yy, int xx, float cs, float sn) {
  float c = 0.5f * (float)(k - 1);
  float ys = (float)yy - c, xs = (float)xx - c;
  float sy = cs * ys - sn * xs + c;
  float sx = sn * ys + cs * xs + c;
  int y0 = (int)floorf(sy), x0 = (int)floorf(sx);
  float wy = sy - (float)y0, wx = sx - (float)x0;
  float acc = 0.f;
#pragma unroll
  for (int dy = 0; dy < 2; ++dy)
#pragma unroll
    for (int dx = 0; dx < 2; ++dx) {
      int yi = y0 + dy, xi = x0 + dx;
      if (yi >= 0 && yi < k && xi >= 0 && xi < k) {
        float w = (dy ? wy : 1.f - wy) * (dx ? wx : 1.f - wx);
        acc += src[yi * k + xi] * w;
      }
    }
  return acc;
}

// lift weights: w [16][3][7][7] -> wrot rows [co*8+m][160], col = ci*49 + yy*7 + xx
__global__ void rot_lift(const float* __restrict__ w, float* __restrict__ wrot) {
  int total = 8 * 16 * 3 * 49;
  for (int idx = blockIdx.x * blockDim.x + threadIdx.x; idx < total;
       idx += gridDim.x * blockDim.x) {
    int xx = idx % 7; int t = idx / 7;
    int yy = t % 7;  t /= 7;
    int ci = t % 3;  t /= 3;
    int co = t % 16; int m = t / 16;
    float th = 2.f * PI_F * (float)m / 8.f;
    float cs = cosf(th), sn = sinf(th);
    const float* src = w + ((size_t)co * 3 + ci) * 49;
    wrot[(size_t)(co * 8 + m) * 160 + ci * 49 + yy * 7 + xx] =
        rotate_sample(src, 7, yy, xx, cs, sn);
  }
}

// gconv weights: w [16][16][8][5][5] -> wrot rows [co*8+m][3200], col = (ci*8+o)*25 + yy*5 + xx
// orientation shift: source orientation = (o - m) mod 8, then rotate by theta_m
__global__ void rot_gconv(const float* __restrict__ w, float* __restrict__ wrot) {
  int total = 8 * 16 * 16 * 8 * 25;
  for (int idx = blockIdx.x * blockDim.x + threadIdx.x; idx < total;
       idx += gridDim.x * blockDim.x) {
    int xx = idx % 5; int t = idx / 5;
    int yy = t % 5;  t /= 5;
    int o  = t % 8;  t /= 8;
    int ci = t % 16; t /= 16;
    int co = t % 16; int m = t / 16;
    float th = 2.f * PI_F * (float)m / 8.f;
    float cs = cosf(th), sn = sinf(th);
    const float* src = w + (((size_t)co * 16 + ci) * 8 + ((o - m) & 7)) * 25;
    wrot[(size_t)(co * 8 + m) * 3200 + (ci * 8 + o) * 25 + yy * 5 + xx] =
        rotate_sample(src, 5, yy, xx, cs, sn);
  }
}

// conv4 1x1 weights (rotation of 1x1 is identity): w4 [16][16][8]
__global__ void shift_w4(const float* __restrict__ w, float* __restrict__ wrot) {
  int total = 8 * 16 * 16 * 8;
  for (int idx = blockIdx.x * blockDim.x + threadIdx.x; idx < total;
       idx += gridDim.x * blockDim.x) {
    int o  = idx % 8;  int t = idx / 8;
    int ci = t % 16;  t /= 16;
    int co = t % 16;  int m = t / 16;
    wrot[(size_t)(co * 8 + m) * 128 + ci * 8 + o] =
        w[((size_t)co * 16 + ci) * 8 + ((o - m) & 7)];
  }
}

// ---------------- pack weights into WMMA A-fragment order (bf16 pairs) ----------------
// A-fragment (16x32 bf16, wave32): lane l -> row M = l%16, h = l/16;
// VGPR v, elem e: K = (v/4)*16 + h*8 + (v%4)*2 + e
// packed index = (((tile*nChunks + chunk)*32 + lane)*8 + v)
__global__ void pack_a(const float* __restrict__ wrot, u32* __restrict__ wpack,
                       int KtotPad, int nChunks) {
  int total = 8 * nChunks * 256;
  for (int idx = blockIdx.x * blockDim.x + threadIdx.x; idx < total;
       idx += gridDim.x * blockDim.x) {
    int v    = idx & 7;
    int lane = (idx >> 3) & 31;
    int rest = idx >> 8;
    int ch   = rest % nChunks;
    int t    = rest / nChunks;
    int row  = t * 16 + (lane & 15);
    int k    = ch * 32 + (v >> 2) * 16 + ((lane >> 4) << 3) + ((v & 3) << 1);
    float f0 = wrot[(size_t)row * KtotPad + k];
    float f1 = wrot[(size_t)row * KtotPad + k + 1];
    wpack[idx] = (u32)f2bf(f0) | ((u32)f2bf(f1) << 16);
  }
}

// ---------------- implicit-GEMM conv via WMMA bf16, 128(M) x 128(N) per block ----------------
// 256 threads = 8 waves; wave w owns output rows [16w,16w+16); block owns 128 spatial cols
// (one contiguous x-run of a single image row: 128-aligned, never crosses a row).
// Double-buffered LDS: stage chunk ch+1 while the 8 WMMAs consume chunk ch (1 barrier/chunk).
// LDS buffer: 32(K) x 128(N) patch tile in B-fragment order: lds[n*32 + k] (bf16).
// B-fragment (32x16 bf16): lane l -> N = l%16, K = (v/4)*16 + (l/16)*8 + (v%4)*2 + e => two 16B runs per lane.
// Each wave loads ONE A fragment per chunk and issues 8 independent WMMAs (8 accumulators).
__global__ void __launch_bounds__(256)
conv_wmma(const u16* __restrict__ xpad, const u32* __restrict__ wpack,
          float* __restrict__ out, int Cin, int kh, int kw, int Hp, int Wp,
          int Ktot, int nChunks) {
  __shared__ __align__(16) u16 ldsB[2 * 128 * 32];   // 2 x 8 KB
  const int tid  = threadIdx.x;
  const int wv   = tid >> 5;
  const int lane = tid & 31;
  const int nBase = blockIdx.x * 128;            // 128 output pixels per block
  const int b    = nBase >> 16;                  // 65536 pixels per image
  const int pix0 = nBase & 65535;
  const int y0   = pix0 >> 8;
  const int x0   = pix0 & 255;
  const int kk2  = kh * kw;
  // staging role: this thread owns K-slot kSt for all 128 columns in groups of 8
  const int kSt  = tid & 31;
  const int nSub = tid >> 5;                     // 0..7

  v8f acc[8];
#pragma unroll
  for (int i = 0; i < 8; ++i) acc[i] = (v8f){0.f,0.f,0.f,0.f,0.f,0.f,0.f,0.f};

  // ---- stage chunk 0 into buffer 0 ----
  {
    u16* buf = ldsB;
    int kg = kSt;
    if (kg < Ktot) {
      int c  = kg / kk2;  int r = kg - c * kk2;
      int dy = r / kw;    int dx = r - dy * kw;
      const u16* row = xpad + (((size_t)b * Cin + c) * Hp + (y0 + dy)) * Wp + (x0 + dx);
#pragma unroll
      for (int j = 0; j < 16; ++j) buf[(j * 8 + nSub) * 32 + kSt] = row[j * 8 + nSub];
    } else {
#pragma unroll
      for (int j = 0; j < 16; ++j) buf[(j * 8 + nSub) * 32 + kSt] = 0;
    }
  }
  __syncthreads();

  const int bo = (lane & 15) * 16 + ((lane >> 4) << 2);   // uint offset within a 16-col tile

  for (int ch = 0; ch < nChunks; ++ch) {
    // ---- stage chunk ch+1 into the other buffer (overlaps with WMMAs below) ----
    if (ch + 1 < nChunks) {
      u16* buf = ldsB + ((ch + 1) & 1) * (128 * 32);
      int kg = (ch + 1) * 32 + kSt;
      if (kg < Ktot) {
        int c  = kg / kk2;  int r = kg - c * kk2;
        int dy = r / kw;    int dx = r - dy * kw;
        const u16* row = xpad + (((size_t)b * Cin + c) * Hp + (y0 + dy)) * Wp + (x0 + dx);
#pragma unroll
        for (int j = 0; j < 16; ++j) buf[(j * 8 + nSub) * 32 + kSt] = row[j * 8 + nSub];
      } else {
#pragma unroll
        for (int j = 0; j < 16; ++j) buf[(j * 8 + nSub) * 32 + kSt] = 0;
      }
      // keep the next A-fragment line hot (global_prefetch_b8)
      __builtin_prefetch((const void*)(wpack +
          ((((size_t)wv * nChunks + (ch + 1)) * 32 + lane) << 3)), 0, 1);
    }
    // ---- one A fragment, eight B fragments, eight WMMAs ----
    union U { u32 u[8]; v16bf v; };
    U A;
    const u32* Ap = wpack + ((((size_t)wv * nChunks + ch) * 32 + lane) << 3);
#pragma unroll
    for (int i = 0; i < 8; ++i) A.u[i] = Ap[i];
    const u32* Lp = (const u32*)(ldsB + (ch & 1) * (128 * 32));
#pragma unroll
    for (int nt = 0; nt < 8; ++nt) {
      U B;
#pragma unroll
      for (int i = 0; i < 8; ++i) B.u[i] = Lp[nt * 256 + bo + (i >> 2) * 8 + (i & 3)];
      acc[nt] = __builtin_amdgcn_wmma_f32_16x16x32_bf16(false, A.v, false, B.v,
                                                        (short)0, acc[nt], false, false);
    }
    __syncthreads();   // staging of ch+1 done for all; reads of buffer ch done
  }

  // ---- write D: lane -> N = lane%16 (+16*nt); VGPR r -> M = r + 8*(lane/16); wave offset 16*wv
  const int nl    = lane & 15;
  const int mbase = wv * 16 + ((lane >> 4) << 3);
  float* outBase = out + (((size_t)b * COUT + mbase) << 16) + pix0 + nl;
#pragma unroll
  for (int r = 0; r < 8; ++r) {
    float* o = outBase + ((size_t)r << 16);
#pragma unroll
    for (int nt = 0; nt < 8; ++nt) o[nt * 16] = acc[nt][r];
  }
}

// ---------------- deterministic BN statistics (per co, over b,m,h,w) ----------------
__global__ void bn_stats1(const float* __restrict__ act, float* __restrict__ part) {
  // grid = 16 co * 64 slices; 1,048,576 elements per co; 16384 per block; 64 per thread
  int co = blockIdx.x >> 6;
  int slice = blockIdx.x & 63;
  float s = 0.f, sq = 0.f;
  int base = slice * 16384 + threadIdx.x;
#pragma unroll 4
  for (int i = 0; i < 64; ++i) {
    int e = base + i * 256;
    int pl = e >> 16; int pix = e & 65535;
    int b = pl >> 3, m = pl & 7;
    float v = act[(((size_t)b * COUT + co * 8 + m) << 16) + pix];
    s += v; sq += v * v;
  }
  __shared__ float ls[256], lq[256];
  ls[threadIdx.x] = s; lq[threadIdx.x] = sq;
  __syncthreads();
  for (int st = 128; st > 0; st >>= 1) {
    if (threadIdx.x < st) {
      ls[threadIdx.x] += ls[threadIdx.x + st];
      lq[threadIdx.x] += lq[threadIdx.x + st];
    }
    __syncthreads();
  }
  if (threadIdx.x == 0) {
    part[blockIdx.x * 2]     = ls[0];
    part[blockIdx.x * 2 + 1] = lq[0];
  }
}

__global__ void bn_stats2(const float* __restrict__ part, float* __restrict__ st2) {
  int co = threadIdx.x;
  if (co < 16) {
    float s = 0.f, sq = 0.f;
    for (int i = 0; i < 64; ++i) {        // fixed order -> deterministic
      s  += part[(co * 64 + i) * 2];
      sq += part[(co * 64 + i) * 2 + 1];
    }
    float mean = s * (1.f / 1048576.f);
    float var  = sq * (1.f / 1048576.f) - mean * mean;
    st2[2 * co]     = mean;
    st2[2 * co + 1] = rsqrtf(var + 1e-5f);
  }
}

// ---------------- BN+ReLU + max over orientation + 1x1 head + sigmoid ----------------
__global__ void final_head(const float* __restrict__ act, const float* __restrict__ st2,
                           const float* __restrict__ g, const float* __restrict__ be,
                           const float* __restrict__ fw, float* __restrict__ out) {
  int idx = blockIdx.x * 256 + threadIdx.x;    // 131072 pixels total
  int b = idx >> 16, pix = idx & 65535;
  float s = 0.f;
#pragma unroll
  for (int co = 0; co < 16; ++co) {
    float mean = st2[2 * co], inv = st2[2 * co + 1];
    float ga = g[co], bb = be[co];
    float mx = 0.f;                            // relu outputs are >= 0
#pragma unroll
    for (int m = 0; m < 8; ++m) {
      float v = act[(((size_t)b * COUT + co * 8 + m) << 16) + pix];
      v = ga * (v - mean) * inv + bb;
      v = fmaxf(v, 0.f);
      mx = fmaxf(mx, v);
    }
    s += fw[co] * mx;
  }
  out[idx] = 1.f / (1.f + expf(-s));
}

// ================================================================================
extern "C" void kernel_launch(void* const* d_in, const int* in_sizes, int n_in,
                              void* d_out, int out_size, void* d_ws, size_t ws_size,
                              hipStream_t stream) {
  const float* x      = (const float*)d_in[0];
  const float* lift_w = (const float*)d_in[1];
  const float* wL[3]  = {(const float*)d_in[2], (const float*)d_in[3], (const float*)d_in[4]};
  const float* w4     = (const float*)d_in[5];
  const float* fw     = (const float*)d_in[6];
  const float* gm[5]  = {(const float*)d_in[7],  (const float*)d_in[9],  (const float*)d_in[11],
                         (const float*)d_in[13], (const float*)d_in[15]};
  const float* bt[5]  = {(const float*)d_in[8],  (const float*)d_in[10], (const float*)d_in[12],
                         (const float*)d_in[14], (const float*)d_in[16]};

  // workspace carve
  char* p = (char*)d_ws;
  auto carve = [&](size_t bytes) -> char* {
    char* r = p; p += (bytes + 255) & ~(size_t)255; return r;
  };
  float* ACT   = (float*)carve((size_t)2 * 128 * 65536 * 4);      // 64 MiB f32 activations
  u16*   PADB  = (u16*)  carve((size_t)2 * 128 * 260 * 260 * 2);  // bf16 padded input
  float* WROT  = (float*)carve((size_t)128 * 3200 * 4);           // rotated weight matrix
  u32*   WPACK = (u32*)  carve((size_t)8 * 100 * 256 * 4);        // WMMA-packed weights
  float* PART  = (float*)carve((size_t)16 * 64 * 2 * 4);
  float* ST    = (float*)carve((size_t)16 * 2 * 4);

  // ---- lift: pad(3) -> 7x7 rotated conv (K=147, padded to 160) ----
  { int total = 2 * 3 * 262 * 262;
    pad_bn_relu_bf16<<<(total + 255) / 256, 256, 0, stream>>>(
        x, PADB, 3, 256, 256, 3, nullptr, nullptr, nullptr, 0, total); }
  hipMemsetAsync(WROT, 0, (size_t)128 * 160 * 4, stream);
  { int total = 8 * 16 * 3 * 49;
    rot_lift<<<(total + 255) / 256, 256, 0, stream>>>(lift_w, WROT); }
  { int total = 8 * 5 * 256;
    pack_a<<<(total + 255) / 256, 256, 0, stream>>>(WROT, WPACK, 160, 5); }
  conv_wmma<<<1024, 256, 0, stream>>>(PADB, WPACK, ACT, 3, 7, 7, 262, 262, 147, 5);
  bn_stats1<<<1024, 256, 0, stream>>>(ACT, PART);
  bn_stats2<<<1, 64, 0, stream>>>(PART, ST);
  { int total = 2 * 128 * 260 * 260;
    pad_bn_relu_bf16<<<(total + 255) / 256, 256, 0, stream>>>(
        ACT, PADB, 128, 256, 256, 2, ST, gm[0], bt[0], 1, total); }

  // ---- three 5x5 g-convs (K = 128*25 = 3200, 100 chunks) ----
  for (int L = 0; L < 3; ++L) {
    { int total = 8 * 16 * 16 * 8 * 25;
      rot_gconv<<<(total + 255) / 256, 256, 0, stream>>>(wL[L], WROT); }
    { int total = 8 * 100 * 256;
      pack_a<<<(total + 255) / 256, 256, 0, stream>>>(WROT, WPACK, 3200, 100); }
    conv_wmma<<<1024, 256, 0, stream>>>(PADB, WPACK, ACT, 128, 5, 5, 260, 260, 3200, 100);
    bn_stats1<<<1024, 256, 0, stream>>>(ACT, PART);
    bn_stats2<<<1, 64, 0, stream>>>(PART, ST);
    int pn = (L < 2) ? 2 : 0;                     // conv4 input is unpadded
    int total = 2 * 128 * (256 + 2 * pn) * (256 + 2 * pn);
    pad_bn_relu_bf16<<<(total + 255) / 256, 256, 0, stream>>>(
        ACT, PADB, 128, 256, 256, pn, ST, gm[L + 1], bt[L + 1], 1, total);
  }

  // ---- conv4: 1x1 g-conv (K = 128, 4 chunks) ----
  { int total = 8 * 16 * 16 * 8;
    shift_w4<<<(total + 255) / 256, 256, 0, stream>>>(w4, WROT); }
  { int total = 8 * 4 * 256;
    pack_a<<<(total + 255) / 256, 256, 0, stream>>>(WROT, WPACK, 128, 4); }
  conv_wmma<<<1024, 256, 0, stream>>>(PADB, WPACK, ACT, 128, 1, 1, 256, 256, 128, 4);
  bn_stats1<<<1024, 256, 0, stream>>>(ACT, PART);
  bn_stats2<<<1, 64, 0, stream>>>(PART, ST);

  // ---- BN + ReLU + orientation max + 1x1 head + sigmoid ----
  final_head<<<512, 256, 0, stream>>>(ACT, ST, gm[4], bt[4], fw, (float*)d_out);
}
